// Attention_70712341561415
// MI455X (gfx1250) — compile-verified, matrix operand order checked by reading, not run on hardware
//
#include <hip/hip_runtime.h>
#include <math.h>

#define CC 256
#define NN 4096
#define BB 4

typedef __attribute__((ext_vector_type(16))) __bf16 v16bf;
typedef __attribute__((ext_vector_type(8)))  float  v8f;
typedef __attribute__((ext_vector_type(4)))  unsigned int u32x4;
typedef __attribute__((ext_vector_type(8)))  int i32x8;
typedef __attribute__((ext_vector_type(4)))  int i32x4;

struct U32B { uint4 lo, hi; };

// 16 contiguous bf16 (as u16) -> WMMA fragment via 2x b128 loads.
__device__ __forceinline__ v16bf load16bf(const unsigned short* p) {
    U32B u;
    u.lo = *(const uint4*)(p);
    u.hi = *(const uint4*)(p + 8);
    return __builtin_bit_cast(v16bf, u);
}

__device__ __forceinline__ unsigned short f2bf(float x) {
    unsigned int u = __builtin_bit_cast(unsigned int, x);
    u += 0x7FFFu + ((u >> 16) & 1u);   // round-to-nearest-even
    return (unsigned short)(u >> 16);
}

__device__ __forceinline__ v8f wmma_bf16(v16bf a, v16bf b, v8f c) {
    return __builtin_amdgcn_wmma_f32_16x16x32_bf16(false, a, false, b, (short)0, c, false, false);
}

// ---------------------------------------------------------------------------
// TDM: async 2D tile load Global->LDS (Tensor Data Mover, TENSORcnt tracked).
// data_size = 4B units. Optional LDS row padding (pad_int_code/pad_amt_code
// per D# group1 encoding: interval 0=2dw..7=256dw, amount 0=1dw..127=128dw).
// ---------------------------------------------------------------------------
__device__ __forceinline__ void tdm_load_2d(unsigned lds_byte_off, const void* gaddr,
                                            unsigned tile_w_dw, unsigned tile_h,
                                            unsigned stride_dw,
                                            unsigned pad_en, unsigned pad_int_code,
                                            unsigned pad_amt_code) {
    unsigned long long ga = (unsigned long long)(size_t)gaddr;
    u32x4 g0;
    g0[0] = 1u;                                              // count=1 (valid), user mode
    g0[1] = lds_byte_off;                                    // lds_addr
    g0[2] = (unsigned)(ga & 0xFFFFFFFFu);                    // global_addr[31:0]
    g0[3] = (unsigned)((ga >> 32) & 0x01FFFFFFu) | (2u << 30); // global_addr[56:32] | type=2
    i32x8 g1;
    g1[0] = (int)((2u << 16) | (pad_en << 20) |
                  (pad_int_code << 22) | (pad_amt_code << 25)); // data_size=4B, padding
    g1[1] = (int)((tile_w_dw & 0xFFFFu) << 16);              // tensor_dim0[15:0]
    g1[2] = (int)(((tile_w_dw >> 16) & 0xFFFFu) | ((tile_h & 0xFFFFu) << 16));
    g1[3] = (int)(((tile_h >> 16) & 0xFFFFu) | ((tile_w_dw & 0xFFFFu) << 16)); // tile_dim0
    g1[4] = (int)(tile_h & 0xFFFFu);                         // tile_dim1 (tile_dim2=0)
    g1[5] = (int)stride_dw;                                  // tensor_dim0_stride[31:0]
    g1[6] = 0;
    g1[7] = 0;
    i32x4 z4 = {0, 0, 0, 0};
#if __clang_major__ >= 23
    i32x8 z8 = {0, 0, 0, 0, 0, 0, 0, 0};
    __builtin_amdgcn_tensor_load_to_lds(g0, g1, z4, z4, z8, 0);
#else
    __builtin_amdgcn_tensor_load_to_lds(g0, g1, z4, z4, 0);
#endif
}

// ---------------------------------------------------------------------------
// Convert the four 256x256 f32 weight matrices to bf16 once.
// ---------------------------------------------------------------------------
__global__ __launch_bounds__(256) void wcvt(const float* __restrict__ Wf,
                                            const float* __restrict__ Wg,
                                            const float* __restrict__ Wh,
                                            const float* __restrict__ Wv,
                                            unsigned short* __restrict__ out) {
    int i = blockIdx.x * 256 + threadIdx.x;
    out[0 * 65536 + i] = f2bf(Wf[i]);
    out[1 * 65536 + i] = f2bf(Wg[i]);
    out[2 * 65536 + i] = f2bf(Wh[i]);
    out[3 * 65536 + i] = f2bf(Wv[i]);
}

// ---------------------------------------------------------------------------
// Transpose + convert x: f32 [C,N] -> bf16 xT [N,C], per batch. 64x64 tiles.
// ---------------------------------------------------------------------------
__global__ __launch_bounds__(256) void xcvt(const float* __restrict__ X,
                                            unsigned short* __restrict__ XT) {
    __shared__ unsigned short T[64 * 72];      // [n][c], row stride 72 u16 (144B)
    const int tid = threadIdx.x;
    const int b = blockIdx.z;
    const int n0 = blockIdx.x * 64, c0 = blockIdx.y * 64;
    const float* Xb = X + (size_t)b * CC * NN;
    unsigned short* XTb = XT + (size_t)b * NN * CC;

    const int r = tid >> 2;             // row within tile (c for load, n for store)
    const int s16 = (tid & 3) * 16;     // 16-wide segment
#pragma unroll
    for (int m = 0; m < 4; ++m) {
        float4 v = *(const float4*)&Xb[(size_t)(c0 + r) * NN + n0 + s16 + m * 4];
        T[(s16 + m * 4 + 0) * 72 + r] = f2bf(v.x);
        T[(s16 + m * 4 + 1) * 72 + r] = f2bf(v.y);
        T[(s16 + m * 4 + 2) * 72 + r] = f2bf(v.z);
        T[(s16 + m * 4 + 3) * 72 + r] = f2bf(v.w);
    }
    __syncthreads();
#pragma unroll
    for (int u = 0; u < 2; ++u) {
        uint4 v = *(const uint4*)&T[r * 72 + s16 + u * 8];
        *(uint4*)&XTb[(size_t)(n0 + r) * CC + c0 + s16 + u * 8] = v;
    }
}

// ---------------------------------------------------------------------------
// out[o,n] = sum_c W[o,c] * XT[n,c] + bias[o]    (one batch per blockIdx.z)
// 256 threads = 8 waves; block tile [256 o x 32 n]; no LDS, no barriers;
// A/B fragments double-buffered in registers so loads overlap WMMA.
// mode 0: bf16 out transposed [N,C]; 1: bf16 [C,N]; 2: f32 [C,N].
// ---------------------------------------------------------------------------
__global__ __launch_bounds__(256) void proj_gemm(
    const unsigned short* __restrict__ XT,      // [N,C] bf16 per batch
    const unsigned short* __restrict__ Wbf,     // [256,256] bf16
    const float* __restrict__ bias,
    unsigned short* __restrict__ outb,
    float* __restrict__ outf,
    int mode)
{
    const int tid = threadIdx.x;
    const int lane = tid & 31, wave = tid >> 5;
    const int l15 = lane & 15, khalf = lane >> 4;
    const int batch = blockIdx.z;
    const int n0 = blockIdx.x * 32;
    const int obase = (wave & 3) * 64;
    const int nloc = ((wave >> 2) << 4) + l15;
    const int nglob = n0 + nloc;

    const unsigned short* xrow = XT + (size_t)batch * NN * CC + (size_t)nglob * CC;
    const unsigned short* wrow[4];
#pragma unroll
    for (int t = 0; t < 4; ++t)
        wrow[t] = &Wbf[(size_t)(obase + t * 16 + l15) * CC];

    v8f acc[4];
#pragma unroll
    for (int t = 0; t < 4; ++t)
#pragma unroll
        for (int r = 0; r < 8; ++r) acc[t][r] = 0.f;

    v16bf bfrag = load16bf(&xrow[khalf * 16]);
    v16bf afrag[4];
#pragma unroll
    for (int t = 0; t < 4; ++t) afrag[t] = load16bf(&wrow[t][khalf * 16]);

#pragma unroll
    for (int k0 = 0; k0 < CC; k0 += 32) {
        v16bf bn = bfrag, an[4];
#pragma unroll
        for (int t = 0; t < 4; ++t) an[t] = afrag[t];
        if (k0 + 32 < CC) {
            bn = load16bf(&xrow[k0 + 32 + khalf * 16]);
#pragma unroll
            for (int t = 0; t < 4; ++t) an[t] = load16bf(&wrow[t][k0 + 32 + khalf * 16]);
        }
#pragma unroll
        for (int t = 0; t < 4; ++t) acc[t] = wmma_bf16(afrag[t], bfrag, acc[t]);
        bfrag = bn;
#pragma unroll
        for (int t = 0; t < 4; ++t) afrag[t] = an[t];
    }

#pragma unroll
    for (int t = 0; t < 4; ++t) {
        const int ob = obase + t * 16 + khalf * 8;
        float4 b0 = *(const float4*)&bias[ob];
        float4 b1 = *(const float4*)&bias[ob + 4];
        acc[t][0] += b0.x; acc[t][1] += b0.y; acc[t][2] += b0.z; acc[t][3] += b0.w;
        acc[t][4] += b1.x; acc[t][5] += b1.y; acc[t][6] += b1.z; acc[t][7] += b1.w;

        if (mode == 0) {
            uint4 pk;
            pk.x = (unsigned)f2bf(acc[t][0]) | ((unsigned)f2bf(acc[t][1]) << 16);
            pk.y = (unsigned)f2bf(acc[t][2]) | ((unsigned)f2bf(acc[t][3]) << 16);
            pk.z = (unsigned)f2bf(acc[t][4]) | ((unsigned)f2bf(acc[t][5]) << 16);
            pk.w = (unsigned)f2bf(acc[t][6]) | ((unsigned)f2bf(acc[t][7]) << 16);
            *(uint4*)&outb[(size_t)batch * NN * CC + (size_t)nglob * CC + ob] = pk;
        } else if (mode == 1) {
#pragma unroll
            for (int r = 0; r < 8; ++r)
                outb[(size_t)batch * CC * NN + (size_t)(ob + r) * NN + nglob] = f2bf(acc[t][r]);
        } else {
#pragma unroll
            for (int r = 0; r < 8; ++r)
                outf[(size_t)batch * CC * NN + (size_t)(ob + r) * NN + nglob] = acc[t][r];
        }
    }
}

// ---------------------------------------------------------------------------
// Flash attention: out[c,j] = sum_i softmax_i(f_i . g_j) * h[c,i]
// g j-tile staged to LDS by the Tensor Data Mover with row padding
// (512B rows -> 528B LDS rows = conflict-free B-fragment reads).
// ---------------------------------------------------------------------------
#define GS_STRIDE 264   // u16 per Gs row: 256 + 8 (TDM pad: 4 dwords / 128-dword row)
#define PT_STRIDE 72    // u16 per PT row

__global__ __launch_bounds__(256) void flash_attn(
    const unsigned short* __restrict__ fT,   // [B][N][C] bf16
    const unsigned short* __restrict__ gT,   // [B][N][C] bf16
    const unsigned short* __restrict__ Hm,   // [B][C][N] bf16
    unsigned short* __restrict__ OT)         // [B][N][C] bf16 (transposed out)
{
    __shared__ float          S[64 * 64];               // 16 KB scores
    __shared__ unsigned short PT[64 * PT_STRIDE];       // P bf16, [j][i]
    __shared__ unsigned short Gs[64 * GS_STRIDE];       // g j-tile [jloc][c] (TDM)
    __shared__ float pmax[256], psum[256];
    __shared__ float m_s[64], l_s[64], alpha_s[64];

    const int tid = threadIdx.x;
    const int lane = tid & 31, wave = tid >> 5;
    const int l15 = lane & 15, khalf = lane >> 4;
    const int b = blockIdx.y;
    const int j0 = blockIdx.x * 64;
    const size_t nc = (size_t)NN * CC;
    const unsigned short* fTb = fT + (size_t)b * nc;
    const unsigned short* Hb  = Hm + (size_t)b * nc;

    const int it = wave & 3, jp = wave >> 2;   // stage-1 ownership
    const int cbase = wave * 32;               // stage-2 ownership

    // --- TDM: pull g[j0..j0+64) rows (64 x 128 dwords) into LDS, padded rows
    if (wave == 0) {
        tdm_load_2d((unsigned)(size_t)&Gs[0],
                    gT + (size_t)b * nc + (size_t)j0 * CC,
                    /*tile_w_dw=*/128, /*tile_h=*/64, /*stride_dw=*/128,
                    /*pad_en=*/1, /*pad_int=*/6 /*128 dw*/, /*pad_amt=*/3 /*4 dw*/);
        __builtin_amdgcn_s_wait_tensorcnt(0);
    }
    if (tid < 64) { m_s[tid] = -3.0e38f; l_s[tid] = 0.f; }

    v8f Oacc[8];
#pragma unroll
    for (int a = 0; a < 8; ++a)
#pragma unroll
        for (int r = 0; r < 8; ++r) Oacc[a][r] = 0.f;

    __syncthreads();

    const unsigned gr0 = ((jp * 2 + 0) * 16 + l15) * GS_STRIDE;
    const unsigned gr1 = ((jp * 2 + 1) * 16 + l15) * GS_STRIDE;

    for (int i0 = 0; i0 < NN; i0 += 64) {
        // ---- stage 1: S = fT x g (double-buffered fragments) ----
        v8f s0, s1;
#pragma unroll
        for (int r = 0; r < 8; ++r) { s0[r] = 0.f; s1[r] = 0.f; }
        const unsigned short* fp = &fTb[(size_t)(i0 + it * 16 + l15) * CC];
        if (i0 + 64 < NN) {
            __builtin_prefetch(fp + 64 * CC, 0, 0);
            __builtin_prefetch(&Hb[(size_t)(cbase + l15) * NN + i0 + 64], 0, 0);
        }
        v16bf a  = load16bf(&fp[khalf * 16]);
        v16bf b0 = load16bf(&Gs[gr0 + khalf * 16]);
        v16bf b1 = load16bf(&Gs[gr1 + khalf * 16]);
#pragma unroll
        for (int k0 = 0; k0 < CC; k0 += 32) {
            v16bf an = a, bn0 = b0, bn1 = b1;
            if (k0 + 32 < CC) {
                an  = load16bf(&fp[k0 + 32 + khalf * 16]);
                bn0 = load16bf(&Gs[gr0 + k0 + 32 + khalf * 16]);
                bn1 = load16bf(&Gs[gr1 + k0 + 32 + khalf * 16]);
            }
            s0 = wmma_bf16(a, b0, s0);
            s1 = wmma_bf16(a, b1, s1);
            a = an; b0 = bn0; b1 = bn1;
        }
#pragma unroll
        for (int r = 0; r < 8; ++r) {
            const int iloc = it * 16 + khalf * 8 + r;
            S[iloc * 64 + (jp * 2 + 0) * 16 + l15] = s0[r];
            S[iloc * 64 + (jp * 2 + 1) * 16 + l15] = s1[r];
        }
        __syncthreads();

        // ---- online softmax over i: 4 partials per column ----
        const int j = tid & 63, part = tid >> 6;
        {
            float mx = -3.0e38f;
#pragma unroll
            for (int i = part * 16; i < part * 16 + 16; ++i)
                mx = fmaxf(mx, S[i * 64 + j]);
            pmax[part * 64 + j] = mx;
        }
        __syncthreads();
        {
            float mx = fmaxf(fmaxf(pmax[j], pmax[64 + j]),
                             fmaxf(pmax[128 + j], pmax[192 + j]));
            mx = fmaxf(mx, m_s[j]);
            float ls = 0.f;
#pragma unroll
            for (int i = part * 16; i < part * 16 + 16; ++i) {
                float p = __expf(S[i * 64 + j] - mx);
                ls += p;
                PT[j * PT_STRIDE + i] = f2bf(p);
            }
            psum[part * 64 + j] = ls;
        }
        __syncthreads();
        if (tid < 64) {
            float mo = m_s[j];
            float mx = fmaxf(fmaxf(pmax[j], pmax[64 + j]),
                             fmaxf(pmax[128 + j], pmax[192 + j]));
            mx = fmaxf(mx, mo);
            float al = __expf(mo - mx);
            l_s[j] = l_s[j] * al + psum[j] + psum[64 + j] + psum[128 + j] + psum[192 + j];
            m_s[j] = mx;
            alpha_s[j] = al;
        }
        __syncthreads();

        // ---- rescale accumulator ----
#pragma unroll
        for (int jt = 0; jt < 4; ++jt) {
            const float al = alpha_s[jt * 16 + l15];
#pragma unroll
            for (int c2 = 0; c2 < 2; ++c2)
#pragma unroll
                for (int r = 0; r < 8; ++r) Oacc[c2 * 4 + jt][r] *= al;
        }

        // ---- stage 2: O += h x P  (K = 64) ----
#pragma unroll
        for (int k0 = 0; k0 < 64; k0 += 32) {
            v16bf a0 = load16bf(&Hb[(size_t)(cbase + l15) * NN + i0 + k0 + khalf * 16]);
            v16bf a1 = load16bf(&Hb[(size_t)(cbase + 16 + l15) * NN + i0 + k0 + khalf * 16]);
#pragma unroll
            for (int jt = 0; jt < 4; ++jt) {
                v16bf bp = load16bf(&PT[(jt * 16 + l15) * PT_STRIDE + k0 + khalf * 16]);
                Oacc[0 * 4 + jt] = wmma_bf16(a0, bp, Oacc[0 * 4 + jt]);
                Oacc[1 * 4 + jt] = wmma_bf16(a1, bp, Oacc[1 * 4 + jt]);
            }
        }
        __syncthreads();
    }

    // ---- normalize, store transposed [N,C] with packed b128 stores ----
#pragma unroll
    for (int jt = 0; jt < 4; ++jt) {
        const float linv = 1.0f / l_s[jt * 16 + l15];
        const int j = j0 + jt * 16 + l15;
#pragma unroll
        for (int c2 = 0; c2 < 2; ++c2) {
            const int cb = cbase + c2 * 16 + khalf * 8;
            const v8f o = Oacc[c2 * 4 + jt];
            uint4 pk;
            pk.x = (unsigned)f2bf(o[0] * linv) | ((unsigned)f2bf(o[1] * linv) << 16);
            pk.y = (unsigned)f2bf(o[2] * linv) | ((unsigned)f2bf(o[3] * linv) << 16);
            pk.z = (unsigned)f2bf(o[4] * linv) | ((unsigned)f2bf(o[5] * linv) << 16);
            pk.w = (unsigned)f2bf(o[6] * linv) | ((unsigned)f2bf(o[7] * linv) << 16);
            *(uint4*)&OT[(size_t)b * nc + (size_t)j * CC + cb] = pk;
        }
    }
}

// ---------------------------------------------------------------------------
extern "C" void kernel_launch(void* const* d_in, const int* in_sizes, int n_in,
                              void* d_out, int out_size, void* d_ws, size_t ws_size,
                              hipStream_t stream) {
    const float* x  = (const float*)d_in[0];
    const float* Wf = (const float*)d_in[1];
    const float* bf = (const float*)d_in[2];
    const float* Wg = (const float*)d_in[3];
    const float* bg = (const float*)d_in[4];
    const float* Wh = (const float*)d_in[5];
    const float* bh = (const float*)d_in[6];
    const float* Wv = (const float*)d_in[7];
    const float* bv = (const float*)d_in[8];

    const size_t nc = (size_t)NN * CC;
    unsigned short* Wbf  = (unsigned short*)d_ws;            // 4 * 65536
    unsigned short* xT   = Wbf + 4 * 65536;                  // B*N*C (bf16, transposed)
    unsigned short* fT   = xT + (size_t)BB * nc;
    unsigned short* gTd  = fT + (size_t)BB * nc;
    unsigned short* Hh   = gTd + (size_t)BB * nc;
    unsigned short* fghT = Hh + (size_t)BB * nc;

    wcvt<<<256, 256, 0, stream>>>(Wf, Wg, Wh, Wv, Wbf);
    xcvt<<<dim3(NN / 64, CC / 64, BB), 256, 0, stream>>>(x, xT);

    dim3 gp(NN / 32, 1, BB);
    proj_gemm<<<gp, 256, 0, stream>>>(xT, Wbf + 0 * 65536, bf, fT, nullptr, 0);
    proj_gemm<<<gp, 256, 0, stream>>>(xT, Wbf + 1 * 65536, bg, gTd, nullptr, 0);
    proj_gemm<<<gp, 256, 0, stream>>>(xT, Wbf + 2 * 65536, bh, Hh, nullptr, 1);

    flash_attn<<<dim3(NN / 64, BB), 256, 0, stream>>>(fT, gTd, Hh, fghT);

    proj_gemm<<<gp, 256, 0, stream>>>(fghT, Wbf + 3 * 65536, bv, nullptr,
                                      (float*)d_out, 2);
}
